// GroupedQueryAttention_58780922413899
// MI455X (gfx1250) — matrix operand order repaired and checked by reading, not compiled, not run on hardware
//
#include <hip/hip_runtime.h>
#include <hip/hip_bf16.h>

typedef __attribute__((ext_vector_type(16))) __bf16 v16bf;
typedef __attribute__((ext_vector_type(8)))  __bf16 v8bf;
typedef __attribute__((ext_vector_type(8)))  float  v8f;

namespace {

constexpr int Bc = 4, Tc = 4096, Fc = 512, Gc = 4, GSc = 128;
constexpr int BR = 128;   // query rows per workgroup (8 waves x 16 rows)
constexpr int BK = 64;    // keys per mainloop iteration
constexpr float SM_SCALE = 0.08838834764831843f; // 1/(sqrt(128)+1e-9)

__device__ __forceinline__ v8f zero_v8f() {
  v8f z;
#pragma unroll
  for (int i = 0; i < 8; ++i) z[i] = 0.0f;
  return z;
}

// A fragment (16x32 bf16): lane L -> row = L%16, K = (L/16)*8 + [0..7] and 16+(L/16)*8 + [0..7]
__device__ __forceinline__ v16bf load_a_frag(const __bf16* row, int kbase, int lhalf) {
  v8bf lo = *(const v8bf*)(row + kbase + lhalf * 8);
  v8bf hi = *(const v8bf*)(row + kbase + 16 + lhalf * 8);
  v16bf a;
#pragma unroll
  for (int i = 0; i < 8; ++i) { a[i] = lo[i]; a[8 + i] = hi[i]; }
  return a;
}

// B fragment (32x16 bf16): lane L -> col = L%16, K = (L/16)*8 + [0..7] and 16+(L/16)*8 + [0..7]; p = base + (L/16)*8
__device__ __forceinline__ v16bf load_b_frag(const __bf16* p) {
  v8bf lo = *(const v8bf*)(p);
  v8bf hi = *(const v8bf*)(p + 16);
  v16bf b;
#pragma unroll
  for (int i = 0; i < 8; ++i) { b[i] = lo[i]; b[8 + i] = hi[i]; }
  return b;
}

// ---------------------------------------------------------------------------
// Kernel 1: per-group QKV projection, f32 in -> bf16 head-major [B][G][T][128]
// Q is pre-scaled by 1/sqrt(d) so the attention mainloop never scales S.
// ---------------------------------------------------------------------------
__global__ __launch_bounds__(256) void gqa_qkv_kernel(
    const float* __restrict__ x,
    const float* __restrict__ Wq, const float* __restrict__ bq,
    const float* __restrict__ Wk, const float* __restrict__ bk,
    const float* __restrict__ Wv, const float* __restrict__ bv,
    __bf16* __restrict__ qo, __bf16* __restrict__ ko, __bf16* __restrict__ vo)
{
  const int tb = blockIdx.x, g = blockIdx.y, b = blockIdx.z;
  const int tid = threadIdx.x;
  const int wave = tid >> 5, lane = tid & 31;
  const int l16 = lane & 15, lhalf = lane >> 4;

  __shared__ __bf16 Wt[128][136];   // Wt[n][k] = W[k][n], padded stride for bank spread

  // Load this wave's 16 rows of X as bf16 A-fragments (reused for all 3 projections)
  const int row = tb * BR + wave * 16 + l16;
  const float* xrow = x + ((size_t)b * Tc + row) * Fc + g * GSc;
  v16bf afrag[4];
#pragma unroll
  for (int kc = 0; kc < 4; ++kc) {
    const float4 u0 = *(const float4*)(xrow + kc * 32 + lhalf * 8);
    const float4 u1 = *(const float4*)(xrow + kc * 32 + lhalf * 8 + 4);
    const float4 u2 = *(const float4*)(xrow + kc * 32 + 16 + lhalf * 8);
    const float4 u3 = *(const float4*)(xrow + kc * 32 + 16 + lhalf * 8 + 4);
    v16bf a;
    a[0]  = (__bf16)u0.x; a[1]  = (__bf16)u0.y; a[2]  = (__bf16)u0.z; a[3]  = (__bf16)u0.w;
    a[4]  = (__bf16)u1.x; a[5]  = (__bf16)u1.y; a[6]  = (__bf16)u1.z; a[7]  = (__bf16)u1.w;
    a[8]  = (__bf16)u2.x; a[9]  = (__bf16)u2.y; a[10] = (__bf16)u2.z; a[11] = (__bf16)u2.w;
    a[12] = (__bf16)u3.x; a[13] = (__bf16)u3.y; a[14] = (__bf16)u3.z; a[15] = (__bf16)u3.w;
    afrag[kc] = a;
  }

  const float* Ws[3] = {Wq, Wk, Wv};
  const float* bs[3] = {bq, bk, bv};
  __bf16* outs[3] = {qo, ko, vo};
  const size_t headoff = ((size_t)(b * Gc + g)) * Tc * GSc;

  for (int p = 0; p < 3; ++p) {
    __syncthreads();   // previous projection finished reading Wt
    const float* W = Ws[p] + (size_t)g * GSc * GSc;
#pragma unroll
    for (int it = 0; it < 16; ++it) {
      int id = tid + it * 256;             // 0..4095
      int kk = id >> 5;                    // 0..127
      int n0 = (id & 31) * 4;
      float4 w4 = *(const float4*)(W + kk * GSc + n0);  // coalesced 16B
      Wt[n0 + 0][kk] = (__bf16)w4.x;
      Wt[n0 + 1][kk] = (__bf16)w4.y;
      Wt[n0 + 2][kk] = (__bf16)w4.z;
      Wt[n0 + 3][kk] = (__bf16)w4.w;
    }
    __syncthreads();

    __bf16* dst = outs[p] + headoff;
    const float* bias = bs[p] + g * GSc;
    const float oscale = (p == 0) ? SM_SCALE : 1.0f;  // fold softmax scale into Q
#pragma unroll
    for (int nt = 0; nt < 8; ++nt) {
      v8f acc = zero_v8f();
#pragma unroll
      for (int kc = 0; kc < 4; ++kc) {
        v16bf bf = load_b_frag(&Wt[nt * 16 + l16][kc * 32 + lhalf * 8]);
        acc = __builtin_amdgcn_wmma_f32_16x16x32_bf16(false, afrag[kc], false, bf,
                                                      (short)0, acc, false, false);
      }
      const float bb = bias[nt * 16 + l16];
#pragma unroll
      for (int r = 0; r < 8; ++r) {
        const int grow = tb * BR + wave * 16 + r + 8 * lhalf;
        dst[(size_t)grow * GSc + nt * 16 + l16] = (__bf16)((acc[r] + bb) * oscale);
      }
    }
  }
}

// ---------------------------------------------------------------------------
// Kernel 2: flash attention per (b,g) head, 128 query rows per workgroup.
// K/V staging is software-pipelined: tile j+1 lives in VGPRs while tile j is
// consumed from LDS, so global-load latency overlaps the WMMA phase; tile j+2
// is prefetched into L2 with global_prefetch_b8.
// ---------------------------------------------------------------------------
__global__ __launch_bounds__(256) void gqa_attn_kernel(
    const __bf16* __restrict__ q, const __bf16* __restrict__ k,
    const __bf16* __restrict__ v, float* __restrict__ out)
{
  const int qb = blockIdx.x, g = blockIdx.y, b = blockIdx.z;
  const int tid = threadIdx.x;
  const int wave = tid >> 5, lane = tid & 31;
  const int l16 = lane & 15, lhalf = lane >> 4;

  __shared__ __bf16 Kt[BK][136];        // keys row-major [key][d], padded
  __shared__ __bf16 Vt[128][BK + 8];    // values transposed [d][key], padded
  __shared__ __bf16 Pb[8][16][BK];      // per-wave probability tile (row-major)

  const size_t headoff = ((size_t)(b * Gc + g)) * Tc * GSc;
  const __bf16* qh = q + headoff;
  const __bf16* kh = k + headoff;
  const __bf16* vh = v + headoff;

  // Q A-fragments stay in registers for the whole pass (pre-scaled by 1/sqrt(d))
  const int qrow = qb * BR + wave * 16 + l16;
  v16bf qfrag[4];
#pragma unroll
  for (int kc = 0; kc < 4; ++kc)
    qfrag[kc] = load_a_frag(qh + (size_t)qrow * GSc, kc * 32, lhalf);

  float rowmax[8], rowsum[8];
  v8f acc[8];  // O tile: 16 rows x 128 cols f32
#pragma unroll
  for (int r = 0; r < 8; ++r) { rowmax[r] = -3.0e38f; rowsum[r] = 0.0f; }
#pragma unroll
  for (int n = 0; n < 8; ++n) acc[n] = zero_v8f();

  // Per-thread staging assignment: 4 chunks of 8 bf16 (K row r, feature chunk dc)
  int ldr[4], ldc[4];
#pragma unroll
  for (int it = 0; it < 4; ++it) {
    const int id = tid + it * 256;        // 0..1023
    ldr[it] = id >> 4;                    // key row 0..63
    ldc[it] = (id & 15) * 8;              // feature chunk
  }

  // Prologue: pull tile 0 into registers
  v8bf kreg[4], vreg[4];
#pragma unroll
  for (int it = 0; it < 4; ++it) {
    const size_t goff = (size_t)ldr[it] * GSc + ldc[it];
    kreg[it] = *(const v8bf*)(kh + goff);
    vreg[it] = *(const v8bf*)(vh + goff);
  }

  for (int j0 = 0; j0 < Tc; j0 += BK) {
    __syncthreads();   // everyone done reading Kt/Vt from previous iteration
    // Commit the in-register tile to LDS (s_wait_loadcnt lands here, not before compute)
#pragma unroll
    for (int it = 0; it < 4; ++it) {
      *(v8bf*)(&Kt[ldr[it]][ldc[it]]) = kreg[it];
      const v8bf vv = vreg[it];
#pragma unroll
      for (int i = 0; i < 8; ++i) Vt[ldc[it] + i][ldr[it]] = vv[i];  // transpose into LDS
    }
    __syncthreads();

    // Issue global loads for the next tile; latency overlaps the compute below
    if (j0 + BK < Tc) {
#pragma unroll
      for (int it = 0; it < 4; ++it) {
        const size_t goff = (size_t)(j0 + BK + ldr[it]) * GSc + ldc[it];
        kreg[it] = *(const v8bf*)(kh + goff);
        vreg[it] = *(const v8bf*)(vh + goff);
      }
    }
    // Warm L2 one tile deeper (global_prefetch_b8)
    if (j0 + 2 * BK < Tc) {
      const size_t poff = (size_t)(j0 + 2 * BK + ldr[0]) * GSc + ldc[0];
      __builtin_prefetch(kh + poff, 0, 3);
      __builtin_prefetch(vh + poff, 0, 3);
    }

    // S = Q * K^T for this wave's 16 rows x 64 keys (Q already scaled)
    v8f st[4];
#pragma unroll
    for (int jt = 0; jt < 4; ++jt) {
      v8f s = zero_v8f();
#pragma unroll
      for (int kc = 0; kc < 4; ++kc) {
        v16bf bf = load_b_frag(&Kt[jt * 16 + l16][kc * 32 + lhalf * 8]);
        s = __builtin_amdgcn_wmma_f32_16x16x32_bf16(false, qfrag[kc], false, bf,
                                                    (short)0, s, false, false);
      }
      st[jt] = s;
    }

    // Online softmax: row m lives across 16 lanes of one half-wave (slot r -> row r + 8*lhalf)
#pragma unroll
    for (int r = 0; r < 8; ++r) {
      float m = fmaxf(fmaxf(st[0][r], st[1][r]), fmaxf(st[2][r], st[3][r]));
      m = fmaxf(m, __shfl_xor(m, 1));
      m = fmaxf(m, __shfl_xor(m, 2));
      m = fmaxf(m, __shfl_xor(m, 4));
      m = fmaxf(m, __shfl_xor(m, 8));
      const float mnew = fmaxf(rowmax[r], m);
      const float alpha = __expf(rowmax[r] - mnew);
      rowmax[r] = mnew;
      float ps = 0.0f;
#pragma unroll
      for (int jt = 0; jt < 4; ++jt) {
        const float pv = __expf(st[jt][r] - mnew);
        st[jt][r] = pv;
        ps += pv;
      }
      ps += __shfl_xor(ps, 1);
      ps += __shfl_xor(ps, 2);
      ps += __shfl_xor(ps, 4);
      ps += __shfl_xor(ps, 8);
      rowsum[r] = rowsum[r] * alpha + ps;
#pragma unroll
      for (int n = 0; n < 8; ++n) acc[n][r] *= alpha;
    }

    // Re-shape P (C layout) -> A layout via wave-private LDS tile
#pragma unroll
    for (int jt = 0; jt < 4; ++jt)
#pragma unroll
      for (int r = 0; r < 8; ++r)
        Pb[wave][r + 8 * lhalf][jt * 16 + l16] = (__bf16)st[jt][r];
    // wave-private region: in-wave DS ordering + compiler dscnt waits suffice

    // O += P * V
#pragma unroll
    for (int kc = 0; kc < 2; ++kc) {
      v16bf pf = load_a_frag(&Pb[wave][l16][0], kc * 32, lhalf);
#pragma unroll
      for (int n = 0; n < 8; ++n) {
        v16bf vf = load_b_frag(&Vt[n * 16 + l16][kc * 32 + lhalf * 8]);
        acc[n] = __builtin_amdgcn_wmma_f32_16x16x32_bf16(false, pf, false, vf,
                                                         (short)0, acc[n], false, false);
      }
    }
  }

  // Normalize and write out f32: out[b][t][g*128 + d]
  float* ob = out + ((size_t)b * Tc) * Fc + g * GSc;
#pragma unroll
  for (int r = 0; r < 8; ++r) {
    const float inv = 1.0f / rowsum[r];
    const int grow = qb * BR + wave * 16 + r + 8 * lhalf;
#pragma unroll
    for (int n = 0; n < 8; ++n)
      ob[(size_t)grow * Fc + n * 16 + l16] = acc[n][r] * inv;
  }
}

} // namespace

extern "C" void kernel_launch(void* const* d_in, const int* in_sizes, int n_in,
                              void* d_out, int out_size, void* d_ws, size_t ws_size,
                              hipStream_t stream) {
  const float* x  = (const float*)d_in[0];
  const float* Wq = (const float*)d_in[1];
  const float* bq = (const float*)d_in[2];
  const float* Wk = (const float*)d_in[3];
  const float* bk = (const float*)d_in[4];
  const float* Wv = (const float*)d_in[5];
  const float* bv = (const float*)d_in[6];
  float* out = (float*)d_out;

  // bf16 Q/K/V head-major [B][G][T][128] in workspace: 3 x 16 MB = 48 MB
  const size_t per = (size_t)Bc * Gc * Tc * GSc;
  __bf16* qws = (__bf16*)d_ws;
  __bf16* kws = qws + per;
  __bf16* vws = kws + per;

  dim3 grid(Tc / BR, Gc, Bc);   // 32 x 4 x 4 = 512 workgroups
  gqa_qkv_kernel<<<grid, 256, 0, stream>>>(x, Wq, bq, Wk, bk, Wv, bv, qws, kws, vws);
  gqa_attn_kernel<<<grid, 256, 0, stream>>>(qws, kws, vws, out);
}
